// lbp_message_passing_network_67808943669331
// MI455X (gfx1250) — compile-verified, matrix operand order checked by reading, not run on hardware
//
#include <hip/hip_runtime.h>
#include <hip/hip_bf16.h>

typedef __attribute__((ext_vector_type(16))) _Float16 v16h;
typedef __attribute__((ext_vector_type(8)))  float    v8f;

#define TPB 256
#define WAVES_PER_BLOCK 8   // 256 threads / wave32

__device__ inline float nzf(float x) {
    // neg_inf_to_zero (also handles +inf like the reference's isinf)
    return (x == __builtin_inff() || x == -__builtin_inff()) ? 0.0f : x;
}

__device__ inline float lse2(float a, float b) {
    float m = fmaxf(a, b);
    if (m == -__builtin_inff()) return m;
    return m + logf(expf(a - m) + expf(b - m));
}

// ---------------------------------------------------------------------------
// utility kernels
// ---------------------------------------------------------------------------
__global__ void zero_kernel(float* __restrict__ p, int n) {
    int i = blockIdx.x * blockDim.x + threadIdx.x;
    if (i < n) p[i] = 0.0f;
}

// scatter-add f->v messages into per-variable sums (optionally degrees)
__global__ void seg_sum_kernel(const float* __restrict__ msgs,
                               const int* __restrict__ edge_var,
                               float* __restrict__ sums,
                               float* __restrict__ deg,   // may be nullptr
                               int n_edges) {
    int e = blockIdx.x * blockDim.x + threadIdx.x;
    if (e >= n_edges) return;
    int v = edge_var[e];
    atomicAdd(&sums[2 * v],     msgs[2 * e]);
    atomicAdd(&sums[2 * v + 1], msgs[2 * e + 1]);
    if (deg) atomicAdd(&deg[v], 1.0f);
}

// ---------------------------------------------------------------------------
// variable->factor update with WMMA MLP (16 edges per wave)
//   x   = var_sums[edge_var[e]] - ftov[e]                       (fp32)
//   h   = relu(x @ W1 + b1)      via v_wmma_f32_16x16x32_f16
//   y   = h @ W2 + b2 + x        via v_wmma_f32_16x16x32_f16
//   vtf = lognorm(0.5*y + 0.5*vtf_old)
// ---------------------------------------------------------------------------
__global__ void vtf_wmma_kernel(const float* __restrict__ var_sums,
                                const float* __restrict__ ftov,
                                float* __restrict__ vtf,            // in/out
                                const int* __restrict__ edge_var,
                                const float* __restrict__ w1,       // [2][8]
                                const float* __restrict__ b1,       // [8]
                                const float* __restrict__ w2,       // [8][2]
                                const float* __restrict__ b2,       // [2]
                                int n_edges) {
    __shared__ float tile[WAVES_PER_BLOCK][16][16];

    const int lane = threadIdx.x & 31;
    const int wave = threadIdx.x >> 5;
    const int base = (blockIdx.x * WAVES_PER_BLOCK + wave) * 16;
    const int n16  = lane & 15;          // column index (B/C/D role)
    const int mh   = (lane >> 4) * 8;    // C/D: lanes>=16 hold rows M=8..15
    const bool rowlane = lane < 16;
    const int e = base + lane;           // edge handled by row-lane `lane`
    const bool valid = rowlane && (e < n_edges);

    float x0 = 0.0f, x1 = 0.0f, old0 = 0.0f, old1 = 0.0f;
    if (valid) {
        int v = edge_var[e];
        float f0 = ftov[2 * e], f1 = ftov[2 * e + 1];
        x0 = var_sums[2 * v]     - f0;
        x1 = var_sums[2 * v + 1] - f1;
        old0 = vtf[2 * e];
        old1 = vtf[2 * e + 1];
    }

    // ---- layer 1: A = x (16x32, K=0,1 used), B = W1 (32x16, K=0,1 / N=0..7)
    v16h A = {};
    if (rowlane) { A[0] = (_Float16)x0; A[1] = (_Float16)x1; }
    v16h B = {};
    if (rowlane && n16 < 8) {
        B[0] = (_Float16)w1[n16];        // W1[0][n]
        B[1] = (_Float16)w1[8 + n16];    // W1[1][n]
    }
    v8f C = {};
    {
        float bias = (n16 < 8) ? b1[n16] : 0.0f;   // all lane halves carry rows
        for (int r = 0; r < 8; ++r) C[r] = bias;
    }
    v8f D = __builtin_amdgcn_wmma_f32_16x16x32_f16(false, A, false, B,
                                                   (short)0, C, false, false);

    // transpose D (row-striped-over-lanes) -> per-edge rows via LDS
    for (int r = 0; r < 8; ++r) tile[wave][mh + r][n16] = D[r];
    __syncthreads();

    // ---- layer 2: A2 = relu(h) (K=0..7), B2 = W2 (K=0..7 / N=0,1)
    v16h A2 = {};
    if (rowlane) {
        for (int j = 0; j < 8; ++j) {
            float h = tile[wave][lane][j];
            A2[j] = (_Float16)(h > 0.0f ? h : 0.0f);
        }
    }
    v16h B2 = {};
    if (rowlane && n16 < 2) {
        for (int j = 0; j < 8; ++j) B2[j] = (_Float16)w2[j * 2 + n16];
    }
    v8f C2 = {};
    {
        float bias = (n16 < 2) ? b2[n16] : 0.0f;
        for (int r = 0; r < 8; ++r) C2[r] = bias;
    }
    __syncthreads();   // everyone done reading tile before overwrite
    v8f D2 = __builtin_amdgcn_wmma_f32_16x16x32_f16(false, A2, false, B2,
                                                    (short)0, C2, false, false);
    for (int r = 0; r < 8; ++r) tile[wave][mh + r][n16] = D2[r];
    __syncthreads();

    if (valid) {
        float y0 = tile[wave][lane][0] + x0;   // residual
        float y1 = tile[wave][lane][1] + x1;
        float t0 = 0.5f * y0 + 0.5f * old0;    // damping
        float t1 = 0.5f * y1 + 0.5f * old1;
        float l = lse2(t0, t1);                // log-normalize
        vtf[2 * e]     = t0 - l;
        vtf[2 * e + 1] = t1 - l;
    }
}

// ---------------------------------------------------------------------------
// factor->variable update (per factor, scalar fp32)
// ---------------------------------------------------------------------------
__global__ void ftov_kernel(const float* __restrict__ pot,
                            const float* __restrict__ vtf,
                            float* __restrict__ ftov,   // in/out
                            int n_factors) {
    int f = blockIdx.x * blockDim.x + threadIdx.x;
    if (f >= n_factors) return;
    float p00 = pot[4 * f], p01 = pot[4 * f + 1];
    float p10 = pot[4 * f + 2], p11 = pot[4 * f + 3];
    float a0 = vtf[4 * f],     a1 = vtf[4 * f + 1];   // position 0 message
    float c0 = vtf[4 * f + 2], c1 = vtf[4 * f + 3];   // position 1 message
    // m0[a] = lse_b(pot[a][b] + vtf1[b])  -> edge 2f
    float m00 = lse2(p00 + c0, p01 + c1);
    float m01 = lse2(p10 + c0, p11 + c1);
    // m1[b] = lse_a(pot[a][b] + vtf0[a])  -> edge 2f+1
    float m10 = lse2(p00 + a0, p10 + a1);
    float m11 = lse2(p01 + a0, p11 + a1);
    float o00 = ftov[4 * f],     o01 = ftov[4 * f + 1];
    float o10 = ftov[4 * f + 2], o11 = ftov[4 * f + 3];
    float t00 = 0.5f * m00 + 0.5f * o00, t01 = 0.5f * m01 + 0.5f * o01;
    float t10 = 0.5f * m10 + 0.5f * o10, t11 = 0.5f * m11 + 0.5f * o11;
    float l0 = lse2(t00, t01), l1 = lse2(t10, t11);
    ftov[4 * f]     = t00 - l0;
    ftov[4 * f + 1] = t01 - l0;
    ftov[4 * f + 2] = t10 - l1;
    ftov[4 * f + 3] = t11 - l1;
}

// ---------------------------------------------------------------------------
// Bethe free energy pieces (two-stage deterministic reduction)
// ---------------------------------------------------------------------------
__device__ inline void block_reduce_store(float v, float* partials) {
    __shared__ float smem[TPB];
    int t = threadIdx.x;
    smem[t] = v;
    __syncthreads();
    for (int s = TPB / 2; s > 0; s >>= 1) {
        if (t < s) smem[t] += smem[t + s];
        __syncthreads();
    }
    if (t == 0) partials[blockIdx.x] = smem[0];
}

// per-factor: sum_ab p_ab * (nz(fb_ab) - nz(pot_ab))   [= p*fb part of
//   (avg_energy - bethe_entropy) restricted to factors]
__global__ void factor_bethe_kernel(const float* __restrict__ pot,
                                    const float* __restrict__ vtf,
                                    float* __restrict__ partials,
                                    int n_factors) {
    int f = blockIdx.x * blockDim.x + threadIdx.x;
    float c = 0.0f;
    if (f < n_factors) {
        float p00 = pot[4 * f], p01 = pot[4 * f + 1];
        float p10 = pot[4 * f + 2], p11 = pot[4 * f + 3];
        float a0 = vtf[4 * f],     a1 = vtf[4 * f + 1];
        float b0 = vtf[4 * f + 2], b1v = vtf[4 * f + 3];
        float f00 = p00 + a0 + b0,  f01 = p01 + a0 + b1v;
        float f10 = p10 + a1 + b0,  f11 = p11 + a1 + b1v;
        float mx = fmaxf(fmaxf(f00, f01), fmaxf(f10, f11));
        float Z = mx + logf(expf(f00 - mx) + expf(f01 - mx) +
                            expf(f10 - mx) + expf(f11 - mx));
        f00 -= Z; f01 -= Z; f10 -= Z; f11 -= Z;
        c = expf(f00) * (nzf(f00) - nzf(p00))
          + expf(f01) * (nzf(f01) - nzf(p01))
          + expf(f10) * (nzf(f10) - nzf(p10))
          + expf(f11) * (nzf(f11) - nzf(p11));
    }
    block_reduce_store(c, partials);
}

// per-variable: -(deg-1) * sum_j exp(vb_j)*nz(vb_j)
__global__ void var_bethe_kernel(const float* __restrict__ sums,
                                 const float* __restrict__ deg,
                                 float* __restrict__ partials,
                                 int n_vars) {
    int i = blockIdx.x * blockDim.x + threadIdx.x;
    float c = 0.0f;
    if (i < n_vars) {
        float l0 = sums[2 * i], l1 = sums[2 * i + 1];
        float Z = lse2(l0, l1);
        float v0 = l0 - Z, v1 = l1 - Z;
        float inner = expf(v0) * nzf(v0) + expf(v1) * nzf(v1);
        c = -(deg[i] - 1.0f) * inner;
    }
    block_reduce_store(c, partials);
}

__global__ void final_reduce_kernel(const float* __restrict__ partials, int n,
                                    float* __restrict__ out) {
    __shared__ double smem[TPB];
    double acc = 0.0;
    for (int i = threadIdx.x; i < n; i += blockDim.x) acc += (double)partials[i];
    int t = threadIdx.x;
    smem[t] = acc;
    __syncthreads();
    for (int s = TPB / 2; s > 0; s >>= 1) {
        if (t < s) smem[t] += smem[t + s];
        __syncthreads();
    }
    if (t == 0) out[0] = (float)smem[0];
}

// ---------------------------------------------------------------------------
// launcher
// ---------------------------------------------------------------------------
extern "C" void kernel_launch(void* const* d_in, const int* in_sizes, int n_in,
                              void* d_out, int out_size, void* d_ws, size_t ws_size,
                              hipStream_t stream) {
    const float* pot   = (const float*)d_in[0];
    const int*   edge  = (const int*)  d_in[1];
    const float* vtf0  = (const float*)d_in[2];
    const float* ftov0 = (const float*)d_in[3];
    const float* w1    = (const float*)d_in[4];   // [ITERS][2][8]
    const float* b1    = (const float*)d_in[5];   // [ITERS][8]
    const float* w2    = (const float*)d_in[6];   // [ITERS][8][2]
    const float* b2    = (const float*)d_in[7];   // [ITERS][2]

    const int n_edges   = in_sizes[1];
    const int n_factors = n_edges / 2;
    const int n_vars    = 1000000;                // per reference constant
    const int iters     = in_sizes[5] / 8;        // b1 = [ITERS][HID=8]

    float* vtf      = (float*)d_ws;                            // n_edges*2
    float* ftov     = vtf      + (size_t)n_edges * 2;          // n_edges*2
    float* var_sums = ftov     + (size_t)n_edges * 2;          // n_vars*2
    float* var_deg  = var_sums + (size_t)n_vars * 2;           // n_vars
    float* partials = var_deg  + (size_t)n_vars;

    hipMemcpyAsync(vtf,  vtf0,  (size_t)n_edges * 2 * sizeof(float),
                   hipMemcpyDeviceToDevice, stream);
    hipMemcpyAsync(ftov, ftov0, (size_t)n_edges * 2 * sizeof(float),
                   hipMemcpyDeviceToDevice, stream);

    const int NB_E   = (n_edges + TPB - 1) / TPB;
    const int NB_F   = (n_factors + TPB - 1) / TPB;
    const int NB_V   = (n_vars + TPB - 1) / TPB;
    const int NB_VTF = (n_edges + 16 * WAVES_PER_BLOCK - 1) / (16 * WAVES_PER_BLOCK);
    const int NB_ZS  = (n_vars * 2 + TPB - 1) / TPB;           // var_sums zero
    const int NB_ZSD = (n_vars * 3 + TPB - 1) / TPB;           // sums+deg zero

    for (int it = 0; it < iters; ++it) {
        zero_kernel<<<NB_ZS, TPB, 0, stream>>>(var_sums, n_vars * 2);
        seg_sum_kernel<<<NB_E, TPB, 0, stream>>>(ftov, edge, var_sums, nullptr,
                                                 n_edges);
        vtf_wmma_kernel<<<NB_VTF, TPB, 0, stream>>>(
            var_sums, ftov, vtf, edge,
            w1 + (size_t)it * 16, b1 + (size_t)it * 8,
            w2 + (size_t)it * 16, b2 + (size_t)it * 2, n_edges);
        ftov_kernel<<<NB_F, TPB, 0, stream>>>(pot, vtf, ftov, n_factors);
    }

    // beliefs + Bethe free energy
    zero_kernel<<<NB_ZSD, TPB, 0, stream>>>(var_sums, n_vars * 3); // sums + deg
    seg_sum_kernel<<<NB_E, TPB, 0, stream>>>(ftov, edge, var_sums, var_deg,
                                             n_edges);
    factor_bethe_kernel<<<NB_F, TPB, 0, stream>>>(pot, vtf, partials, n_factors);
    var_bethe_kernel<<<NB_V, TPB, 0, stream>>>(var_sums, var_deg,
                                               partials + NB_F, n_vars);
    final_reduce_kernel<<<1, TPB, 0, stream>>>(partials, NB_F + NB_V,
                                               (float*)d_out);
}
